// HippocampalProsodyAttention_29480655520332
// MI455X (gfx1250) — matrix operand order repaired and checked by reading, not compiled, not run on hardware
//
#include <hip/hip_runtime.h>

#define B_   2
#define S_   2048
#define H_   1024
#define NH_  16
#define HD_  64
#define MTOT (B_*S_)   // 4096

typedef __attribute__((ext_vector_type(16))) __bf16 v16bf;
typedef __attribute__((ext_vector_type(8)))  __bf16 v8bf;
typedef __attribute__((ext_vector_type(4)))  __bf16 v4bf;
typedef __attribute__((ext_vector_type(8)))  float  v8f;
typedef __attribute__((ext_vector_type(4)))  float  v4f;

// gcc-style int4 to match the async-LDS builtin's parameter type
typedef int v4i_ __attribute__((vector_size(16)));
typedef __attribute__((address_space(1))) v4i_* as1_v4i;
typedef __attribute__((address_space(3))) v4i_* as3_v4i;

#if defined(__has_builtin)
#if __has_builtin(__builtin_amdgcn_global_load_async_to_lds_b128) && \
    __has_builtin(__builtin_amdgcn_s_wait_asynccnt)
#define HAVE_ASYNC_LDS 1
#endif
#endif

__device__ __forceinline__ float sigmoidf_(float x) { return 1.0f / (1.0f + __expf(-x)); }

// Build a 16-element bf16 fragment from two contiguous 8-half (16B) chunks.
__device__ __forceinline__ v16bf load_frag2(const __bf16* p0, const __bf16* p1) {
    v8bf lo = *(const v8bf*)p0;
    v8bf hi = *(const v8bf*)p1;
    v16bf r;
#pragma unroll
    for (int i = 0; i < 8; i++) { r[i] = lo[i]; r[i + 8] = hi[i]; }
    return r;
}
// B fragment: 16 contiguous halfs.
__device__ __forceinline__ v16bf load_frag16(const __bf16* p) { return load_frag2(p, p + 8); }

// -------------------------------------------------------------------------
// float -> bf16 bulk convert (vectorized x4)
// -------------------------------------------------------------------------
__global__ void cvt_bf16_kernel(const float* __restrict__ x, __bf16* __restrict__ y, int n4) {
    int i = blockIdx.x * blockDim.x + threadIdx.x;
    if (i < n4) {
        v4f v = ((const v4f*)x)[i];
        v4bf o;
#pragma unroll
        for (int j = 0; j < 4; j++) o[j] = (__bf16)v[j];
        ((v4bf*)y)[i] = o;
    }
}

// -------------------------------------------------------------------------
// Gates: memscale[row] = 1 + 0.5*sigmoid(hidden@mg_w + mg_b)
//        gains[row][h] = sigmoid(prosody@pg_w + pg_b)
// one wave per row (wave32), lane-parallel dot + shfl reduce
// -------------------------------------------------------------------------
__global__ __launch_bounds__(256)
void gates_kernel(const float* __restrict__ hidden, const float* __restrict__ pros,
                  const float* __restrict__ pg_w, const float* __restrict__ pg_b,
                  const float* __restrict__ mg_w, const float* __restrict__ mg_b,
                  float* __restrict__ gains, float* __restrict__ memscale) {
    int wave = threadIdx.x >> 5, lane = threadIdx.x & 31;
    int row = blockIdx.x * 8 + wave;
    const float* hr = hidden + (size_t)row * H_;
    float s = 0.f;
    for (int j = lane; j < H_; j += 32) s += hr[j] * mg_w[j];
#pragma unroll
    for (int off = 16; off > 0; off >>= 1) s += __shfl_xor(s, off, 32);
    if (lane == 0) memscale[row] = 1.0f + 0.5f * sigmoidf_(s + mg_b[0]);
    if (lane < NH_) {
        float g = pg_b[lane];
#pragma unroll
        for (int j = 0; j < 4; j++) g += pros[row * 4 + j] * pg_w[j * NH_ + lane];
        gains[row * NH_ + lane] = sigmoidf_(g);
    }
}

// -------------------------------------------------------------------------
// GEMM  Y[M,N] = X(bf16)[M,K] @ W(f32)[K,N] + bias, via v_wmma_f32_16x16x32_bf16
// Block: 256 threads = 8 waves (2x4), tile 64x128, each wave 32x32 (2x2 WMMA).
// A tile staged with GLOBAL_LOAD_ASYNC_TO_LDS_B128 (ASYNCcnt) when available.
// W slab for the *next* K-step is prefetched into registers before the WMMAs.
// MODE 0: bf16 out; MODE 1: bf16 out * (1+gain[row][col/64]); MODE 2: f32 out (NT)
// -------------------------------------------------------------------------
template <int MODE>
__global__ __launch_bounds__(256)
void gemm_wmma(const __bf16* __restrict__ X, const float* __restrict__ W,
               const float* __restrict__ bias, const float* __restrict__ gains,
               __bf16* __restrict__ Ybf, float* __restrict__ Yf,
               int M, int N, int K) {
    __shared__ __align__(16) __bf16 As[64 * 40];   // row-major, pad 40 halfs/row
    __shared__ __align__(16) __bf16 Bt[128 * 40];  // W tile transposed [n][k]

    int tid = threadIdx.x;
    int wave = tid >> 5, lane = tid & 31;
    int wm = wave >> 2, wn = wave & 3;
    int ln = lane & 15, hs = lane >> 4;
    int m_base = blockIdx.y * 64, n_base = blockIdx.x * 128;

    v8f acc[2][2] = {};

    // W slab prefetch registers (32x128 f32 / 256 threads = 16 each)
    float wreg[16];
#pragma unroll
    for (int i = 0; i < 16; i++) {
        int idx = i * 256 + tid;
        int kr = idx >> 7, nc = idx & 127;
        wreg[i] = W[(size_t)kr * N + n_base + nc];
    }

    // per-thread A-copy coordinates: one 16B chunk each (64 rows x 4 chunks)
    int ar = tid >> 2, ac = (tid & 3) * 8;

    for (int kk = 0; kk < K; kk += 32) {
        // ---- stage A tile 64x32 bf16 ----
        {
            const __bf16* gp = X + (size_t)(m_base + ar) * K + kk + ac;
            __bf16* lp = &As[ar * 40 + ac];
#ifdef HAVE_ASYNC_LDS
            __builtin_amdgcn_global_load_async_to_lds_b128(
                (as1_v4i)(v4i_*)(void*)const_cast<__bf16*>(gp),
                (as3_v4i)(v4i_*)(void*)lp, 0, 0);
#else
            *(v8bf*)lp = *(const v8bf*)gp;
#endif
        }
        // ---- stage W tile 32x128 transposed into Bt[n][k] from wreg ----
#pragma unroll
        for (int i = 0; i < 16; i++) {
            int idx = i * 256 + tid;
            int kr = idx >> 7, nc = idx & 127;
            Bt[nc * 40 + kr] = (__bf16)wreg[i];
        }
#ifdef HAVE_ASYNC_LDS
        __builtin_amdgcn_s_wait_asynccnt(0);
#endif
        __syncthreads();

        // prefetch next W slab into regs (overlaps with the WMMAs below)
        if (kk + 32 < K) {
#pragma unroll
            for (int i = 0; i < 16; i++) {
                int idx = i * 256 + tid;
                int kr = idx >> 7, nc = idx & 127;
                wreg[i] = W[(size_t)(kk + 32 + kr) * N + n_base + nc];
            }
        }

        v16bf a0 = load_frag2(&As[(wm * 32 + ln) * 40 + 16 * hs],
                              &As[(wm * 32 + ln) * 40 + 8 + 16 * hs]);
        v16bf a1 = load_frag2(&As[(wm * 32 + 16 + ln) * 40 + 16 * hs],
                              &As[(wm * 32 + 16 + ln) * 40 + 8 + 16 * hs]);
        v16bf b0 = load_frag16(&Bt[(wn * 32 + ln) * 40 + 16 * hs]);
        v16bf b1 = load_frag16(&Bt[(wn * 32 + 16 + ln) * 40 + 16 * hs]);

        acc[0][0] = __builtin_amdgcn_wmma_f32_16x16x32_bf16(false, a0, false, b0, (short)0, acc[0][0], false, false);
        acc[0][1] = __builtin_amdgcn_wmma_f32_16x16x32_bf16(false, a0, false, b1, (short)0, acc[0][1], false, false);
        acc[1][0] = __builtin_amdgcn_wmma_f32_16x16x32_bf16(false, a1, false, b0, (short)0, acc[1][0], false, false);
        acc[1][1] = __builtin_amdgcn_wmma_f32_16x16x32_bf16(false, a1, false, b1, (short)0, acc[1][1], false, false);
        __syncthreads();
    }

#pragma unroll
    for (int mi = 0; mi < 2; mi++)
#pragma unroll
        for (int ni = 0; ni < 2; ni++) {
            int col = n_base + wn * 32 + ni * 16 + ln;
#pragma unroll
            for (int r = 0; r < 8; r++) {
                int row = m_base + wm * 32 + mi * 16 + r + 8 * hs;
                float v = acc[mi][ni][r] + bias[col];
                if (MODE == 1) v *= (1.0f + gains[row * NH_ + (col >> 6)]);
                if (MODE == 2) __builtin_nontemporal_store(v, &Yf[(size_t)row * N + col]);
                else           Ybf[(size_t)row * N + col] = (__bf16)v;
            }
        }
}

// -------------------------------------------------------------------------
// Attention: one workgroup per (bh, 16-query strip).
// LDS: scores f32 [16][2048] (128KB) + P bf16 [16][2064] (64.5KB)
//      + Vt bf16 [64][272] (34KB)  => ~227KB of the WGP's 320KB.
// Phase2: K-fragment double buffering hides global latency behind WMMA.
// Phase3: float4-vectorized softmax; attn written with non-temporal b128.
// Phase4: ctx = P @ V, K-blocked with transposed V staging; per-wave K split.
// Phase5: cross-wave LDS reduction, write ctx bf16.
// -------------------------------------------------------------------------
__global__ __launch_bounds__(256)
void attn_wmma(const __bf16* __restrict__ q, const __bf16* __restrict__ k,
               const __bf16* __restrict__ v, const float* __restrict__ memscale,
               float* __restrict__ attn, __bf16* __restrict__ ctx) {
    extern __shared__ __align__(16) char smem_raw[];
    float*  sc = (float*)smem_raw;                                   // 16*2048 f32
    __bf16* P  = (__bf16*)(smem_raw + 16 * 2048 * 4);                // 16*2064 bf16
    __bf16* Vt = (__bf16*)(smem_raw + 16 * 2048 * 4 + 16 * 2064 * 2);// 64*272 bf16

    int tid = threadIdx.x, wave = tid >> 5, lane = tid & 31;
    int ln = lane & 15, hs = lane >> 4;
    int q0 = blockIdx.x * 16;
    int bh = blockIdx.y;
    int b = bh >> 4, hh = bh & 15;
    const int q_hi = q0 + 15;

    // Q fragments (shared by all of this wave's N tiles), 2 K-steps of 32
    const __bf16* qrow = q + ((size_t)(b * S_) + q0 + ln) * H_ + hh * HD_;
    v16bf qa[2];
#pragma unroll
    for (int ks = 0; ks < 2; ks++)
        qa[ks] = load_frag2(qrow + ks * 32 + 16 * hs, qrow + ks * 32 + 8 + 16 * hs);

    float msc[8];
#pragma unroll
    for (int r = 0; r < 8; r++) msc[r] = memscale[b * S_ + q0 + r + 8 * hs];

    // ---- Phase 2: scores into LDS; software-pipelined K-fragment loads ----
    const __bf16* kbase = k + (size_t)(b * S_) * H_ + hh * HD_;
    int d = q_hi - wave * 256;
    int tmax = d < 0 ? -1 : (d >> 4);
    if (tmax > 15) tmax = 15;

    v16bf nb0, nb1;
    if (tmax >= 0) {
        const __bf16* krow = kbase + (size_t)(wave * 256 + ln) * H_;
        nb0 = load_frag16(krow + 16 * hs);
        nb1 = load_frag16(krow + 32 + 16 * hs);
    }
    for (int t = 0; t <= tmax; t++) {
        v16bf cb0 = nb0, cb1 = nb1;
        if (t < tmax) {
            const __bf16* krow = kbase + (size_t)(wave * 256 + (t + 1) * 16 + ln) * H_;
            nb0 = load_frag16(krow + 16 * hs);
            nb1 = load_frag16(krow + 32 + 16 * hs);
        }
        v8f a = {};
        a = __builtin_amdgcn_wmma_f32_16x16x32_bf16(false, qa[0], false, cb0, (short)0, a, false, false);
        a = __builtin_amdgcn_wmma_f32_16x16x32_bf16(false, qa[1], false, cb1, (short)0, a, false, false);
        int n0 = wave * 256 + t * 16;
#pragma unroll
        for (int r = 0; r < 8; r++) {
            int row = q0 + r + 8 * hs;
            int col = n0 + ln;
            float s = a[r] * 0.125f * msc[r];
            if (col > row) s = -1e30f;
            sc[(r + 8 * hs) * 2048 + col] = s;
        }
    }
    for (int t = tmax < -1 ? 0 : tmax + 1; t < 16; t++) {
        int n0 = wave * 256 + t * 16;
#pragma unroll
        for (int r = 0; r < 8; r++) sc[(r + 8 * hs) * 2048 + n0 + ln] = -1e30f;
    }
    __syncthreads();

    // ---- Phase 3: full-row softmax, float4-vectorized; 2 rows per wave ----
#pragma unroll
    for (int rr = 0; rr < 2; rr++) {
        int row = wave * 2 + rr;
        v4f* srow4 = (v4f*)(sc + row * 2048);     // 512 quads
        float m = -3.0e38f;
        for (int c4 = lane; c4 < 512; c4 += 32) {
            v4f x = srow4[c4];
            m = fmaxf(fmaxf(fmaxf(m, x[0]), fmaxf(x[1], x[2])), x[3]);
        }
#pragma unroll
        for (int off = 16; off > 0; off >>= 1) m = fmaxf(m, __shfl_xor(m, off, 32));
        float l = 0.f;
        for (int c4 = lane; c4 < 512; c4 += 32) {
            v4f x = srow4[c4], e;
#pragma unroll
            for (int j = 0; j < 4; j++) { e[j] = __expf(x[j] - m); l += e[j]; }
            srow4[c4] = e;
        }
#pragma unroll
        for (int off = 16; off > 0; off >>= 1) l += __shfl_xor(l, off, 32);
        float inv = 1.0f / l;
        v4f* arow4 = (v4f*)(attn + ((size_t)bh * S_ + q0 + row) * S_);
        v4bf* prow4 = (v4bf*)(P + row * 2064);
        for (int c4 = lane; c4 < 512; c4 += 32) {
            v4f e = srow4[c4], p;
            v4bf pb;
#pragma unroll
            for (int j = 0; j < 4; j++) { p[j] = e[j] * inv; pb[j] = (__bf16)p[j]; }
            __builtin_nontemporal_store(p, &arow4[c4]);  // attn is write-once > L2
            prow4[c4] = pb;
        }
    }
    __syncthreads();

    // ---- Phase 4: ctx = P @ V, K-blocks of 256, skip fully-masked blocks ----
    v8f pa[4] = {};
    int nblk = (q_hi >> 8) + 1;
    for (int blk = 0; blk < nblk; blk++) {
        // stage V block transposed: Vt[hd][kk]; b64 global reads, 4x b16 LDS stores
        for (int i = tid; i < 64 * 256 / 4; i += 256) {
            int kk = i >> 4, hq = i & 15;           // hq: quad of hd
            const v4bf* gp = (const v4bf*)(v + ((size_t)(b * S_) + blk * 256 + kk) * H_ + hh * HD_ + hq * 4);
            v4bf x = *gp;
            if (blk + 1 < nblk)
                __builtin_prefetch((const void*)(gp + 256 * H_ / 4), 0, 2);
#pragma unroll
            for (int j = 0; j < 4; j++) Vt[(hq * 4 + j) * 272 + kk] = x[j];
        }
        __syncthreads();
        int kloc = wave * 32;  // each wave owns one 32-wide K step per block
        v16bf aP = load_frag2(&P[ln * 2064 + blk * 256 + kloc + 16 * hs],
                              &P[ln * 2064 + blk * 256 + kloc + 8 + 16 * hs]);
#pragma unroll
        for (int nt = 0; nt < 4; nt++) {
            v16bf bV = load_frag16(&Vt[(nt * 16 + ln) * 272 + kloc + 16 * hs]);
            pa[nt] = __builtin_amdgcn_wmma_f32_16x16x32_bf16(false, aP, false, bV, (short)0, pa[nt], false, false);
        }
        __syncthreads();
    }

    // ---- Phase 5: cross-wave reduce (reuse score LDS), write ctx bf16 ----
    float* red = sc;
#pragma unroll
    for (int nt = 0; nt < 4; nt++)
#pragma unroll
        for (int r = 0; r < 8; r++)
            red[((wave * 4 + nt) * 8 + r) * 32 + lane] = pa[nt][r];
    __syncthreads();
    for (int o = tid * 4; o < tid * 4 + 4; o++) {
        int nt = o >> 8, rem = o & 255, r = rem >> 5, lq = rem & 31;
        float s = 0.f;
#pragma unroll
        for (int w = 0; w < 8; w++) s += red[w * 1024 + (nt * 8 + r) * 32 + lq];
        int M = r + 8 * (lq >> 4);
        int hd = nt * 16 + (lq & 15);
        ctx[((size_t)(b * S_) + q0 + M) * H_ + hh * HD_ + hd] = (__bf16)s;
    }
}

// -------------------------------------------------------------------------
extern "C" void kernel_launch(void* const* d_in, const int* in_sizes, int n_in,
                              void* d_out, int out_size, void* d_ws, size_t ws_size,
                              hipStream_t stream) {
    (void)in_sizes; (void)n_in; (void)out_size; (void)ws_size;
    const float* hidden = (const float*)d_in[0];
    const float* pros   = (const float*)d_in[1];
    const float* q_w = (const float*)d_in[2];  const float* q_b = (const float*)d_in[3];
    const float* k_w = (const float*)d_in[4];  const float* k_b = (const float*)d_in[5];
    const float* v_w = (const float*)d_in[6];  const float* v_b = (const float*)d_in[7];
    const float* o_w = (const float*)d_in[8];  const float* o_b = (const float*)d_in[9];
    const float* pg_w = (const float*)d_in[10]; const float* pg_b = (const float*)d_in[11];
    const float* mg_w = (const float*)d_in[12]; const float* mg_b = (const float*)d_in[13];

    float* out  = (float*)d_out;                  // [B,S,H]
    float* attn = out + (size_t)B_ * S_ * H_;     // [B,NH,S,S]

    char* ws = (char*)d_ws;
    size_t off = 0;
    auto alloc = [&](size_t bytes) {
        void* p = ws + off;
        off = (off + bytes + 255) & ~(size_t)255;
        return p;
    };
    __bf16* hb = (__bf16*)alloc((size_t)MTOT * H_ * 2);
    __bf16* qb = (__bf16*)alloc((size_t)MTOT * H_ * 2);
    __bf16* kb = (__bf16*)alloc((size_t)MTOT * H_ * 2);
    __bf16* vb = (__bf16*)alloc((size_t)MTOT * H_ * 2);
    __bf16* cb = (__bf16*)alloc((size_t)MTOT * H_ * 2);
    float* gains    = (float*)alloc((size_t)MTOT * NH_ * 4);
    float* memscale = (float*)alloc((size_t)MTOT * 4);

    int n4 = MTOT * H_ / 4;
    cvt_bf16_kernel<<<(n4 + 255) / 256, 256, 0, stream>>>(hidden, hb, n4);
    gates_kernel<<<MTOT / 8, 256, 0, stream>>>(hidden, pros, pg_w, pg_b, mg_w, mg_b, gains, memscale);

    dim3 gg(H_ / 128, MTOT / 64);
    gemm_wmma<1><<<gg, 256, 0, stream>>>(hb, q_w, q_b, gains, qb, nullptr, MTOT, H_, H_);
    gemm_wmma<0><<<gg, 256, 0, stream>>>(hb, k_w, k_b, nullptr, kb, nullptr, MTOT, H_, H_);
    gemm_wmma<0><<<gg, 256, 0, stream>>>(hb, v_w, v_b, nullptr, vb, nullptr, MTOT, H_, H_);

    size_t shmem = (size_t)16 * 2048 * 4 + (size_t)16 * 2064 * 2 + (size_t)64 * 272 * 2;
    dim3 ga(S_ / 16, B_ * NH_);
    attn_wmma<<<ga, 256, shmem, stream>>>(qb, kb, vb, memscale, attn, cb);

    gemm_wmma<2><<<gg, 256, 0, stream>>>(cb, o_w, o_b, nullptr, nullptr, out, MTOT, H_, H_);
}
